// GraphSAGE_30562987278723
// MI455X (gfx1250) — matrix operand order repaired and checked by reading, not compiled, not run on hardware
//
#include <hip/hip_runtime.h>

// ---------------------------------------------------------------------------
// GraphSAGE-style layer for MI455X (gfx1250, wave32, WMMA + async-to-LDS).
//   h = X@W + b_W ; xi,xj = split(h)
//   m_e = xj[snd[e]] + EF@W_e + b_We ; m_e /= (deg[rcv[e]]+1e-6)
//   out = xi + scatter_add(m_e, rcv)
// GEMMs: split-bf16 (hi+lo) v_wmma_f32_16x16x32_bf16 (3 WMMAs / K-step) for
// ~f32 accuracy at bf16 matrix-core rates. Edge-feature stream (410 MB) is
// double-buffered into LDS with GLOBAL_LOAD_ASYNC_TO_LDS_B128 (ASYNCcnt).
// ---------------------------------------------------------------------------

typedef __attribute__((ext_vector_type(16))) __bf16 v16bf;
typedef __attribute__((ext_vector_type(8)))  float  v8f;

#define DFEAT 128
#define KT    4      // 128 / 32 k-steps
#define AROW  136    // LDS row stride in bf16 elems (272B: 16B aligned, conflict-free)
#define EDGE_BLOCKS 4096

union FragBF { v16bf v; uint4 q[2]; };

// ---- helpers ---------------------------------------------------------------

__global__ void zero_f32_kernel(float* __restrict__ p, int n) {
  int i = blockIdx.x * blockDim.x + threadIdx.x;
  if (i < n) p[i] = 0.0f;
}

__global__ void degree_kernel(const int* __restrict__ recv, float* __restrict__ deg, int n_edges) {
  int i = blockIdx.x * blockDim.x + threadIdx.x;
  if (i < n_edges) atomicAdd(&deg[recv[i]], 1.0f);
}

__global__ void invdeg_kernel(float* __restrict__ deg, int n) {
  int i = blockIdx.x * blockDim.x + threadIdx.x;
  if (i < n) deg[i] = 1.0f / (deg[i] + 1e-6f);
}

static __device__ __forceinline__ unsigned short bf_bits(__bf16 h) {
  union { __bf16 b; unsigned short s; } u; u.b = h; return u.s;
}

// Pack W[k][n] (row-major KxN f32) into WMMA B-fragment order, split hi/lo bf16.
// Fragment element e of lane(half, n) in k-tile kt must be W[kt*32 + half*16 + e][ntile*16+n].
__global__ void pack_b_kernel(const float* __restrict__ Wsrc,
                              unsigned short* __restrict__ hi,
                              unsigned short* __restrict__ lo, int K, int N) {
  int idx = blockIdx.x * blockDim.x + threadIdx.x;
  if (idx >= K * N) return;
  int k = idx / N, n = idx % N;
  float v = Wsrc[idx];
  int pidx = (((n >> 4) * (K >> 5) + (k >> 5)) * 16 + (n & 15)) * 32 + (k & 31);
  __bf16 h = (__bf16)v;
  hi[pidx] = bf_bits(h);
  lo[pidx] = bf_bits((__bf16)(v - (float)h));
}

// ---- B-fragment preload (per wave n-tile, all k-steps, hi+lo) --------------

static __device__ __forceinline__ void load_b_frags(const unsigned short* __restrict__ Bhi,
                                                    const unsigned short* __restrict__ Blo,
                                                    int ntile, FragBF* bh, FragBF* bl) {
  const int lane = threadIdx.x & 31;
  const int hf = lane >> 4, nn = lane & 15;
#pragma unroll
  for (int kt = 0; kt < KT; ++kt) {
    const int bbase = ((ntile * KT + kt) * 16 + nn) * 32 + hf * 16;
    const uint4* pbh = (const uint4*)(Bhi + bbase);
    const uint4* pbl = (const uint4*)(Blo + bbase);
    bh[kt].q[0] = pbh[0]; bh[kt].q[1] = pbh[1];
    bl[kt].q[0] = pbl[0]; bl[kt].q[1] = pbl[1];
  }
}

// ---- core WMMA: 16x16 over K=128, split-bf16, B preloaded ------------------

static __device__ __forceinline__ v8f tile_mma(const __bf16* lds_hi, const __bf16* lds_lo,
                                               const FragBF* bh, const FragBF* bl, v8f c) {
  const int lane = threadIdx.x & 31;
  const int hf = lane >> 4, m = lane & 15;
#pragma unroll
  for (int kt = 0; kt < KT; ++kt) {
    FragBF ah, al;
    // A 16x32 bf16 layout: elems 0..15 -> K = hf*16 + e (matches packed B)
    ah.q[0] = *(const uint4*)(lds_hi + m * AROW + kt * 32 + hf * 16);
    ah.q[1] = *(const uint4*)(lds_hi + m * AROW + kt * 32 + hf * 16 + 8);
    al.q[0] = *(const uint4*)(lds_lo + m * AROW + kt * 32 + hf * 16);
    al.q[1] = *(const uint4*)(lds_lo + m * AROW + kt * 32 + hf * 16 + 8);
    c = __builtin_amdgcn_wmma_f32_16x16x32_bf16(false, ah.v, false, bh[kt].v, (short)0, c, false, false);
    c = __builtin_amdgcn_wmma_f32_16x16x32_bf16(false, ah.v, false, bl[kt].v, (short)0, c, false, false);
    c = __builtin_amdgcn_wmma_f32_16x16x32_bf16(false, al.v, false, bh[kt].v, (short)0, c, false, false);
  }
  return c;
}

// ---- async stage of one 16xDFEAT f32 tile into LDS -------------------------
// Full tiles: each of 256 threads issues 2x GLOBAL_LOAD_ASYNC_TO_LDS_B128
// (16B/lane). Tail tile: guarded synchronous loads. Caller must
// s_wait_asynccnt 0 + __syncthreads() before reading.

static __device__ __forceinline__ void stage_tile(const float* __restrict__ src, int tile,
                                                  int n_rows_total, float* rawbuf, int tid) {
  if ((tile + 1) * 16 <= n_rows_total) {
    const unsigned lds_base = (unsigned)(unsigned long long)(const void*)rawbuf;
    const unsigned long long gbase =
        (unsigned long long)(const void*)src + (unsigned long long)tile * (16u * DFEAT * 4u);
#pragma unroll
    for (int k = 0; k < 2; ++k) {
      unsigned off  = (unsigned)(tid + k * 256) * 16u;   // byte offset within 8KB tile
      unsigned ldsa = lds_base + off;
      asm volatile("global_load_async_to_lds_b128 %0, %1, %2"
                   :: "v"(ldsa), "v"(off), "s"(gbase)
                   : "memory");
    }
  } else {
    for (int i = tid; i < 16 * DFEAT; i += 256) {
      int mm = i >> 7, kk = i & 127;
      int r = tile * 16 + mm;
      rawbuf[mm * DFEAT + kk] = (r < n_rows_total) ? src[r * DFEAT + kk] : 0.0f;
    }
  }
}

// ---- node GEMM: h = X@W + b_W ; xi -> out, xj -> ws ------------------------
// 512 threads = 16 waves; wave w owns n-tile w (N = 256 -> 16 tiles).

__global__ __launch_bounds__(512) void node_gemm_kernel(
    const float* __restrict__ X,
    const unsigned short* __restrict__ Whi, const unsigned short* __restrict__ Wlo,
    const float* __restrict__ bW,
    float* __restrict__ out_xi, float* __restrict__ xj, int n_nodes) {
  __shared__ __align__(16) __bf16 lds_hi[16 * AROW];
  __shared__ __align__(16) __bf16 lds_lo[16 * AROW];
  const int tile = blockIdx.x;
  const int tid  = threadIdx.x;
  for (int i = tid; i < 16 * DFEAT; i += 512) {        // coalesced stage + split
    int mm = i >> 7, kk = i & 127;
    int row = tile * 16 + mm;
    float v = (row < n_nodes) ? X[row * DFEAT + kk] : 0.0f;
    __bf16 h = (__bf16)v;
    lds_hi[mm * AROW + kk] = h;
    lds_lo[mm * AROW + kk] = (__bf16)(v - (float)h);
  }
  const int wave = tid >> 5;                            // = ntile, 0..15
  FragBF bh[KT], bl[KT];
  load_b_frags(Whi, Wlo, wave, bh, bl);
  __syncthreads();
  v8f c = {};
  c = tile_mma(lds_hi, lds_lo, bh, bl, c);
  const int lane = tid & 31, hf = lane >> 4, nn = lane & 15;
  const int col = wave * 16 + nn;                       // 0..255
  const float bias = bW[col];
#pragma unroll
  for (int r = 0; r < 8; ++r) {                         // D layout: VGPR r -> row 8*hf + r
    int row = tile * 16 + hf * 8 + r;
    if (row < n_nodes) {
      float v = c[r] + bias;
      if (col < DFEAT) out_xi[row * DFEAT + col] = v;   // nodes initialized to xi
      else             xj[row * DFEAT + (col - DFEAT)] = v;
    }
  }
}

// ---- edge GEMM + fused gather/normalize/scatter ----------------------------
// Grid-strided multi-tile loop, double-buffered async LDS staging.
// 256 threads = 8 waves; wave w owns n-tile w (N = 128 -> 8 tiles).

__global__ __launch_bounds__(256) void edge_gemm_scatter_kernel(
    const float* __restrict__ EF,
    const unsigned short* __restrict__ Whi, const unsigned short* __restrict__ Wlo,
    const float* __restrict__ bWe, const float* __restrict__ xj,
    const int* __restrict__ senders, const int* __restrict__ receivers,
    const float* __restrict__ invdeg, float* __restrict__ out,
    int n_edges, int n_tiles) {
  __shared__ __align__(16) float  raw[2][16 * DFEAT];   // 2 x 8 KB raw f32 staging
  __shared__ __align__(16) __bf16 lds_hi[16 * AROW];
  __shared__ __align__(16) __bf16 lds_lo[16 * AROW];

  const int tid  = threadIdx.x;
  const int wave = tid >> 5;                            // = ntile, 0..7
  const int lane = tid & 31, hf = lane >> 4, nn = lane & 15;

  int tile = blockIdx.x;
  if (tile >= n_tiles) return;

  // W_e fragments live in registers for the whole block lifetime.
  FragBF bh[KT], bl[KT];
  load_b_frags(Whi, Wlo, wave, bh, bl);
  const int col = wave * 16 + nn;                       // 0..127
  const float bias = bWe[col];

  // prologue: stage first tile
  stage_tile(EF, tile, n_edges, &raw[0][0], tid);
  asm volatile("s_wait_asynccnt 0" ::: "memory");
  __syncthreads();

  int buf = 0;
  for (;;) {
    const int next = tile + (int)gridDim.x;
    if (next < n_tiles)                                 // prefetch next tile (no wait)
      stage_tile(EF, next, n_edges, &raw[buf ^ 1][0], tid);

    // convert current raw f32 tile -> split bf16
    const float* cur = &raw[buf][0];
    for (int i = tid; i < 16 * DFEAT; i += 256) {
      float v = cur[i];
      __bf16 h = (__bf16)v;
      lds_hi[(i >> 7) * AROW + (i & 127)] = h;
      lds_lo[(i >> 7) * AROW + (i & 127)] = (__bf16)(v - (float)h);
    }
    __syncthreads();

    v8f c = {};
    c = tile_mma(lds_hi, lds_lo, bh, bl, c);

#pragma unroll
    for (int r = 0; r < 8; ++r) {
      int e = tile * 16 + hf * 8 + r;
      if (e < n_edges) {
        int s  = senders[e];
        int rc = receivers[e];
        float v = (c[r] + bias + xj[s * DFEAT + col]) * invdeg[rc];
        atomicAdd(&out[rc * DFEAT + col], v);
      }
    }

    if (next >= n_tiles) break;
    asm volatile("s_wait_asynccnt 0" ::: "memory");     // next raw tile landed
    __syncthreads();                                    // all waves done with bf16 bufs
    buf ^= 1;
    tile = next;
  }
}

// ---- launch ----------------------------------------------------------------

extern "C" void kernel_launch(void* const* d_in, const int* in_sizes, int n_in,
                              void* d_out, int out_size, void* d_ws, size_t ws_size,
                              hipStream_t stream) {
  const float* node_features = (const float*)d_in[0];
  const int*   senders       = (const int*)d_in[1];
  const int*   receivers     = (const int*)d_in[2];
  const float* edge_features = (const float*)d_in[3];
  const float* W             = (const float*)d_in[4];
  const float* bW            = (const float*)d_in[5];
  const float* We            = (const float*)d_in[6];
  const float* bWe           = (const float*)d_in[7];
  (void)n_in; (void)out_size; (void)ws_size;

  const int n_nodes = in_sizes[0] / DFEAT;
  const int n_edges = in_sizes[1];

  // workspace layout (256B aligned)
  char* ws = (char*)d_ws;
  size_t off = 0;
  float* xj = (float*)(ws + off);
  off += (size_t)n_nodes * DFEAT * sizeof(float);      off = (off + 255) & ~(size_t)255;
  float* deg = (float*)(ws + off);                      // degree, then invdeg in place
  off += (size_t)n_nodes * sizeof(float);              off = (off + 255) & ~(size_t)255;
  unsigned short* Whi = (unsigned short*)(ws + off);   off += (size_t)DFEAT * 256 * 2; off = (off + 255) & ~(size_t)255;
  unsigned short* Wlo = (unsigned short*)(ws + off);   off += (size_t)DFEAT * 256 * 2; off = (off + 255) & ~(size_t)255;
  unsigned short* Wehi = (unsigned short*)(ws + off);  off += (size_t)DFEAT * DFEAT * 2; off = (off + 255) & ~(size_t)255;
  unsigned short* Welo = (unsigned short*)(ws + off);

  // 1) degrees
  zero_f32_kernel<<<(n_nodes + 255) / 256, 256, 0, stream>>>(deg, n_nodes);
  degree_kernel<<<(n_edges + 255) / 256, 256, 0, stream>>>(receivers, deg, n_edges);
  invdeg_kernel<<<(n_nodes + 255) / 256, 256, 0, stream>>>(deg, n_nodes);

  // 2) pack weights into WMMA B-fragment order (split bf16 hi/lo)
  pack_b_kernel<<<(DFEAT * 256 + 255) / 256, 256, 0, stream>>>(W, Whi, Wlo, DFEAT, 256);
  pack_b_kernel<<<(DFEAT * DFEAT + 255) / 256, 256, 0, stream>>>(We, Wehi, Welo, DFEAT, DFEAT);

  // 3) node projection: out = xi, ws = xj
  const int node_tiles = (n_nodes + 15) / 16;
  node_gemm_kernel<<<node_tiles, 512, 0, stream>>>(node_features, Whi, Wlo, bW,
                                                   (float*)d_out, xj, n_nodes);

  // 4) edge projection + gather + normalize + scatter-add (async double-buffer)
  const int edge_tiles = (n_edges + 15) / 16;
  const int edge_grid  = edge_tiles < EDGE_BLOCKS ? edge_tiles : EDGE_BLOCKS;
  edge_gemm_scatter_kernel<<<edge_grid, 256, 0, stream>>>(edge_features, Wehi, Welo, bWe,
                                                          xj, senders, receivers, deg,
                                                          (float*)d_out, n_edges, edge_tiles);
}
